// FuzzyArcFaceLoss_19928648254288
// MI455X (gfx1250) — matrix operand order repaired and checked by reading, not compiled, not run on hardware
//
#include <hip/hip_runtime.h>
#include <hip/hip_bf16.h>
#include <math.h>

typedef __attribute__((ext_vector_type(8)))  _Float16 v8h;
typedef __attribute__((ext_vector_type(16))) _Float16 v16h;
typedef __attribute__((ext_vector_type(8)))  float    v8f;

#define N_BATCH 2048
#define D_FEAT  512
#define N_CLASS 100000
#define S_SCALE 64.0f
#define MARGIN  0.5f
#define TAU_T   0.1f
#define EPS_A   1e-7f

#define CHUNK_CLASSES 400            // classes per workgroup chunk
#define N_CHUNKS      250            // 100000 / 400
#define TILES_PER_CHUNK 25           // 400 / 16
#define NEG_BIG (-1.0e30f)

#define WAVES_PER_WG 8               // 256 threads, 128 rows per WG
#define ROWS_PER_WG  (WAVES_PER_WG * 16)

// LDS staging: one B tile = 16 classes x 512 K x 2B = 16 KB.
// Pad each class row by 16B (stride 1040) so lane l reads banks 4l..4l+3.
// 4-deep ring buffer -> one barrier per tile is sufficient.
#define B_CLASS_STRIDE 1040
#define B_TILE_BYTES   (16 * B_CLASS_STRIDE)   // 16640
#define N_BUFS 4

// Scheduling-group helper (mask 0x100 = DS read, 0x008 = MFMA/WMMA).
#if __has_builtin(__builtin_amdgcn_sched_group_barrier)
#define SGB(mask, size, id) __builtin_amdgcn_sched_group_barrier(mask, size, id)
#else
#define SGB(mask, size, id)
#endif

// ---------------------------------------------------------------------------
// Kernel 1: L2-normalize rows of a [rows x 512] fp32 matrix into f16.
// ---------------------------------------------------------------------------
__global__ __launch_bounds__(256) void norm_rows_f16(
    const float* __restrict__ in, _Float16* __restrict__ out) {
  __shared__ float red[256];
  const int row = blockIdx.x;
  const int t = threadIdx.x;
  const float* p = in + (size_t)row * D_FEAT;
  float a = p[t];
  float b = p[t + 256];
  red[t] = a * a + b * b;
  __syncthreads();
  for (int o = 128; o > 0; o >>= 1) {
    if (t < o) red[t] += red[t + o];
    __syncthreads();
  }
  const float n = sqrtf(red[0]);
  const float sc = 1.0f / fmaxf(n, 1e-12f);
  _Float16* q = out + (size_t)row * D_FEAT;
  q[t]       = (_Float16)(a * sc);
  q[t + 256] = (_Float16)(b * sc);
}

// ---------------------------------------------------------------------------
// Kernel 2: fused GEMM (f16 WMMA, f32 acc) + ArcFace margin + online
// logsumexp / argmax partials.
//   blockIdx.x = chunk (0..249), blockIdx.y = row group (0..15)
// 8 waves per WG; wave w owns rows (blockIdx.y*8+w)*16 .. +15.
// A is register-resident for the whole kernel (16 x v16h = 128 VGPRs);
// B tiles staged Global->LDS via async-to-LDS, 4-deep ring, 1 barrier/tile;
// B fragments double-buffered in registers, order pinned by sched groups.
// ---------------------------------------------------------------------------
__global__ __launch_bounds__(256) void arcface_gemm_chunk(
    const _Float16* __restrict__ xh, const _Float16* __restrict__ wh,
    const int* __restrict__ label, float* __restrict__ olabel,
    float* __restrict__ pm, float* __restrict__ ps,
    float* __restrict__ pmv, int* __restrict__ pmi) {
  __shared__ __align__(16) char smem[N_BUFS][B_TILE_BYTES];

  const int chunk   = blockIdx.x;
  const int tid     = threadIdx.x;
  const int wid     = tid >> 5;
  const int lane    = tid & 31;
  const int lane15  = lane & 15;
  const bool hi     = lane >= 16;
  const int rowbase = (blockIdx.y * WAVES_PER_WG + wid) * 16;
  const int cstart  = chunk * CHUNK_CLASSES;

  // Labels for the 8 rows this lane's accumulator elements belong to.
  int lab[8];
#pragma unroll
  for (int j = 0; j < 8; ++j)
    lab[j] = label[rowbase + j + (hi ? 8 : 0)];

  // Per-row online logsumexp state + running argmax (per lane).
  float m[8], s[8], mv[8];
  int mi[8];
#pragma unroll
  for (int j = 0; j < 8; ++j) {
    m[j] = NEG_BIG; s[j] = 0.0f; mv[j] = NEG_BIG; mi[j] = 0x7fffffff;
  }

  // A fragments (contiguous k order, same as B): row = rowbase + lane15;
  // lanes 0-15 hold K 0-15, lanes 16-31 hold K 16-31.
  // Load once, keep register-resident for all 25 tiles.
  const _Float16* arow =
      xh + (size_t)(rowbase + lane15) * D_FEAT + (hi ? 16 : 0);
  v16h afrag[16];
#pragma unroll
  for (int k = 0; k < 16; ++k) {
    v8h alo = *(const v8h*)(arow + k * 32);
    v8h ahi = *(const v8h*)(arow + k * 32 + 8);
    afrag[k] = __builtin_shufflevector(alo, ahi, 0, 1, 2, 3, 4, 5, 6, 7,
                                       8, 9, 10, 11, 12, 13, 14, 15);
  }

  // ---- async staging helper: tile `t` of this chunk into buffer `buf` ----
  // 1024 x 16B chunks per tile, 4 per thread; lane-consecutive for coalescing.
  auto stage_tile = [&](int t, int buf) {
    const int cbase = cstart + t * 16;
    const unsigned smbase = (unsigned)(size_t)(&smem[buf][0]);
#pragma unroll
    for (int q = 0; q < 4; ++q) {
      const int u = q * 256 + tid;          // 0..1023
      const int cls = u >> 6;               // 0..15
      const int off16 = u & 63;             // 16B units within a class row
      const char* gp = (const char*)wh +
                       ((size_t)(cbase + cls) << 10) + ((size_t)off16 << 4);
      const unsigned lds = smbase + cls * B_CLASS_STRIDE + (off16 << 4);
      asm volatile("global_load_async_to_lds_b128 %0, %1, off"
                   :: "v"(lds), "v"(gp)
                   : "memory");
    }
  };

  stage_tile(0, 0);   // prologue

  for (int t = 0; t < TILES_PER_CHUNK; ++t) {
    // Prefetch next tile into the ring, then wait for the current tile
    // (async loads complete in order: <=4 outstanding == current tile done).
    // Ring depth 4 guarantees buf[(t+1)&3] is untouched by any wave still
    // computing tile t-1 (the max lag allowed by the collective barrier).
    if (t + 1 < TILES_PER_CHUNK) {
      stage_tile(t + 1, (t + 1) & (N_BUFS - 1));
      asm volatile("s_wait_asynccnt 0x4" ::: "memory");
    } else {
      asm volatile("s_wait_asynccnt 0x0" ::: "memory");
    }
    __syncthreads();  // all waves' async writes now visible in LDS

    // B fragment base in LDS: column n = class lane15 of this tile;
    // lanes 0-15 read K k..k+15, lanes 16-31 read K k+16..k+31.
    const char* bbase =
        &smem[t & (N_BUFS - 1)][0] + lane15 * B_CLASS_STRIDE + (hi ? 32 : 0);

    // Register double-buffered K loop, prefetch distance 2: fragments for
    // k and k+1 are resident; loads for k+2 are issued before WMMA k.
    v8h b0lo = *(const v8h*)(bbase + 0);
    v8h b0hi = *(const v8h*)(bbase + 16);
    v8h b1lo = *(const v8h*)(bbase + 64);
    v8h b1hi = *(const v8h*)(bbase + 80);
    v8f acc = {};
#pragma unroll
    for (int k = 0; k < 16; k += 2) {
      v16h b0 = __builtin_shufflevector(b0lo, b0hi, 0, 1, 2, 3, 4, 5, 6, 7,
                                        8, 9, 10, 11, 12, 13, 14, 15);
      if (k + 2 < 16) {
        b0lo = *(const v8h*)(bbase + (k + 2) * 64);
        b0hi = *(const v8h*)(bbase + (k + 2) * 64 + 16);
      }
      acc = __builtin_amdgcn_wmma_f32_16x16x32_f16(
          false, afrag[k], false, b0, (short)0, acc, false, false);

      v16h b1 = __builtin_shufflevector(b1lo, b1hi, 0, 1, 2, 3, 4, 5, 6, 7,
                                        8, 9, 10, 11, 12, 13, 14, 15);
      if (k + 3 < 16) {
        b1lo = *(const v8h*)(bbase + (k + 3) * 64);
        b1hi = *(const v8h*)(bbase + (k + 3) * 64 + 16);
      }
      acc = __builtin_amdgcn_wmma_f32_16x16x32_f16(
          false, afrag[k + 1], false, b1, (short)0, acc, false, false);
    }

    // Pin the pipeline order for this scheduling region:
    // 4 DS reads, then 14 x (1 WMMA ; 2 DS reads), then 2 WMMA drain.
    SGB(0x100, 4, 0);
#pragma unroll
    for (int g = 0; g < 14; ++g) {
      SGB(0x008, 1, 0);
      SGB(0x100, 2, 0);
    }
    SGB(0x008, 2, 0);

    const int cbase = cstart + t * 16;
    const int c = cbase + lane15;  // class of all 8 elements in this lane
#pragma unroll
    for (int j = 0; j < 8; ++j) {
      float cy = acc[j];
      float val = S_SCALE * cy;
      if (lab[j] == c) {
        // Fuzzy ArcFace margin on the target logit.
        float acy = fabsf(cy);
        float mu = (acy >= TAU_T) ? acy : 1.0f;
        float th = acosf(fminf(fmaxf(cy, -1.0f + EPS_A), 1.0f - EPS_A));
        val = S_SCALE * cosf(th + MARGIN * mu);
        olabel[rowbase + j + (hi ? 8 : 0)] = val;  // output[r, label[r]]
      }
      // Online logsumexp update (finite sentinel keeps exp args sane).
      float nm = fmaxf(m[j], val);
      s[j] = s[j] * __expf(m[j] - nm) + __expf(val - nm);
      m[j] = nm;
      if (val > mv[j] || (val == mv[j] && c < mi[j])) { mv[j] = val; mi[j] = c; }
    }
    // No trailing barrier: 4-deep ring makes buffer reuse safe.
  }

  // Reduce the 16 lanes of each half (xor 1,2,4,8 stays within a half).
#pragma unroll
  for (int j = 0; j < 8; ++j) {
    float mm = m[j], ss = s[j], vv = mv[j];
    int ii = mi[j];
#pragma unroll
    for (int off = 1; off <= 8; off <<= 1) {
      float om = __shfl_xor(mm, off, 32);
      float os = __shfl_xor(ss, off, 32);
      float nm = fmaxf(mm, om);
      ss = ss * __expf(mm - nm) + os * __expf(om - nm);
      mm = nm;
      float ov = __shfl_xor(vv, off, 32);
      int oi = __shfl_xor(ii, off, 32);
      if (ov > vv || (ov == vv && oi < ii)) { vv = ov; ii = oi; }
    }
    if (lane15 == 0) {  // lanes 0 and 16 write their half's row
      const int r = rowbase + j + (hi ? 8 : 0);
      const size_t idx = (size_t)r * N_CHUNKS + chunk;
      pm[idx] = mm; ps[idx] = ss; pmv[idx] = vv; pmi[idx] = ii;
    }
  }
}

// ---------------------------------------------------------------------------
// Kernel 3: combine the 250 per-chunk partials of each row.
// ---------------------------------------------------------------------------
__global__ __launch_bounds__(256) void row_reduce(
    const float* __restrict__ pm, const float* __restrict__ ps,
    const float* __restrict__ pmv, const int* __restrict__ pmi,
    const float* __restrict__ olabel, const int* __restrict__ label,
    float* __restrict__ rowloss, float* __restrict__ rowcorr) {
  __shared__ float sm[256], ssum[256], smv[256];
  __shared__ int smi[256];
  const int row = blockIdx.x;
  const int t = threadIdx.x;
  float m = NEG_BIG, s = 0.0f, v = NEG_BIG;
  int i = 0x7fffffff;
  if (t < N_CHUNKS) {
    const size_t idx = (size_t)row * N_CHUNKS + t;
    m = pm[idx]; s = ps[idx]; v = pmv[idx]; i = pmi[idx];
  }
  sm[t] = m; ssum[t] = s; smv[t] = v; smi[t] = i;
  __syncthreads();
  for (int o = 128; o > 0; o >>= 1) {
    if (t < o) {
      float m2 = sm[t + o], s2 = ssum[t + o];
      float nm = fmaxf(sm[t], m2);
      ssum[t] = ssum[t] * __expf(sm[t] - nm) + s2 * __expf(m2 - nm);
      sm[t] = nm;
      float v2 = smv[t + o];
      int i2 = smi[t + o];
      if (v2 > smv[t] || (v2 == smv[t] && i2 < smi[t])) {
        smv[t] = v2; smi[t] = i2;
      }
    }
    __syncthreads();
  }
  if (t == 0) {
    const float logz = sm[0] + logf(ssum[0]);
    rowloss[row] = logz - olabel[row];
    rowcorr[row] = (smi[0] == label[row]) ? 1.0f : 0.0f;
  }
}

// ---------------------------------------------------------------------------
// Kernel 4: means over the 2048 rows -> (loss, prec1).
// ---------------------------------------------------------------------------
__global__ __launch_bounds__(256) void finalize(
    const float* __restrict__ rowloss, const float* __restrict__ rowcorr,
    float* __restrict__ out) {
  __shared__ float sl[256], sc[256];
  const int t = threadIdx.x;
  float l = 0.0f, c = 0.0f;
  for (int k = t; k < N_BATCH; k += 256) {
    l += rowloss[k];
    c += rowcorr[k];
  }
  sl[t] = l; sc[t] = c;
  __syncthreads();
  for (int o = 128; o > 0; o >>= 1) {
    if (t < o) { sl[t] += sl[t + o]; sc[t] += sc[t + o]; }
    __syncthreads();
  }
  if (t == 0) {
    out[0] = sl[0] / (float)N_BATCH;                 // loss (mean CE)
    out[1] = 100.0f * sc[0] / (float)N_BATCH;        // prec1 (%)
  }
}

// ---------------------------------------------------------------------------
extern "C" void kernel_launch(void* const* d_in, const int* in_sizes, int n_in,
                              void* d_out, int out_size, void* d_ws,
                              size_t ws_size, hipStream_t stream) {
  (void)in_sizes; (void)n_in; (void)out_size; (void)ws_size;
  const float* x      = (const float*)d_in[0];  // [2048, 512]
  const float* weight = (const float*)d_in[1];  // [100000, 512]
  const int*   label  = (const int*)d_in[2];    // [2048]
  float*       out    = (float*)d_out;          // [2] : loss, prec1

  // Workspace layout (256B-aligned slabs).
  char* w = (char*)d_ws;
  size_t off = 0;
  _Float16* xh = (_Float16*)(w + off); off += (size_t)N_BATCH * D_FEAT * 2;
  off = (off + 255) & ~(size_t)255;
  _Float16* wh = (_Float16*)(w + off); off += (size_t)N_CLASS * D_FEAT * 2;
  off = (off + 255) & ~(size_t)255;
  float* olabel = (float*)(w + off); off += (size_t)N_BATCH * 4;
  off = (off + 255) & ~(size_t)255;
  float* pm  = (float*)(w + off); off += (size_t)N_BATCH * N_CHUNKS * 4;
  off = (off + 255) & ~(size_t)255;
  float* ps  = (float*)(w + off); off += (size_t)N_BATCH * N_CHUNKS * 4;
  off = (off + 255) & ~(size_t)255;
  float* pmv = (float*)(w + off); off += (size_t)N_BATCH * N_CHUNKS * 4;
  off = (off + 255) & ~(size_t)255;
  int*   pmi = (int*)(w + off);   off += (size_t)N_BATCH * N_CHUNKS * 4;
  off = (off + 255) & ~(size_t)255;
  float* rowloss = (float*)(w + off); off += (size_t)N_BATCH * 4;
  off = (off + 255) & ~(size_t)255;
  float* rowcorr = (float*)(w + off); off += (size_t)N_BATCH * 4;

  norm_rows_f16<<<N_BATCH, 256, 0, stream>>>(x, xh);
  norm_rows_f16<<<N_CLASS, 256, 0, stream>>>(weight, wh);
  arcface_gemm_chunk<<<dim3(N_CHUNKS, N_BATCH / ROWS_PER_WG), 256, 0, stream>>>(
      xh, wh, label, olabel, pm, ps, pmv, pmi);
  row_reduce<<<N_BATCH, 256, 0, stream>>>(pm, ps, pmv, pmi, olabel, label,
                                          rowloss, rowcorr);
  finalize<<<1, 256, 0, stream>>>(rowloss, rowcorr, out);
}
